// GraphTransformerLayer_23450521436514
// MI455X (gfx1250) — matrix operand order repaired and checked by reading, not compiled, not run on hardware
//
#include <hip/hip_runtime.h>
#include <hip/hip_bf16.h>

typedef __attribute__((ext_vector_type(16))) __bf16 v16bf;
typedef __attribute__((ext_vector_type(8)))  __bf16 v8bf;
typedef __attribute__((ext_vector_type(8)))  float  v8f;

#define DMODEL 128

// ---------------------------------------------------------------- utilities
__global__ __launch_bounds__(256) void zero_f32(float* __restrict__ p, long n) {
    long i = (long)blockIdx.x * 256 + threadIdx.x;
    if (i < n) p[i] = 0.f;
}

__global__ __launch_bounds__(256) void cvt_bf16(const float* __restrict__ s,
                                                __bf16* __restrict__ d, long n) {
    long i = (long)blockIdx.x * 256 + threadIdx.x;
    if (i < n) d[i] = (__bf16)s[i];
}

// dst[c*R + r] = (bf16)src[r*C + c]   (weights -> transposed bf16 [out,in])
__global__ __launch_bounds__(256) void transpose_cvt(const float* __restrict__ s,
                                                     __bf16* __restrict__ d,
                                                     int R, int C) {
    long i = (long)blockIdx.x * 256 + threadIdx.x;
    if (i >= (long)R * C) return;
    int r = (int)(i / C), c = (int)(i % C);
    d[(long)c * R + r] = (__bf16)s[i];
}

// ------------------------------------------------------- WMMA bf16 GEMM
// C[M,Nout] = A[M,Kdim] @ Bt[Nout,Kdim]^T  (+bias +res, relu), f32 and/or bf16 out.
// Wave tile: 16 rows x 64 cols (4 accumulators, A frag reused 4x per K-step).
// A frag layout (contiguous, same as B): lane<16 -> K 0..15, lane>=16 -> K 16..31
// B frag layout (this code's order):       lane<16 -> K 0..15, lane>=16 -> K 16..31 (contiguous)
__global__ __launch_bounds__(256) void gemm_wmma_bf16(
    const __bf16* __restrict__ A, const __bf16* __restrict__ Bt,
    const float* __restrict__ bias, const float* __restrict__ res,
    float* __restrict__ outF, __bf16* __restrict__ outBF,
    int M, int Nout, int Kdim, int relu)
{
    const int lane = threadIdx.x & 31;
    const int wave = threadIdx.x >> 5;
    const int tilesN = Nout >> 6;                       // 64-wide wave tiles
    const long totalTiles = (long)(M >> 4) * tilesN;
    const long tid = (long)blockIdx.x * 8 + wave;       // wave-uniform
    if (tid >= totalTiles) return;                      // whole-wave exit: EXEC all-1s
    const int tm = (int)(tid / tilesN);
    const int tn = (int)(tid % tilesN);

    const int hi16   = lane >> 4;                       // 0 | 1 half-wave group
    const int kbaseA = hi16 << 4;                       // 0 | 16
    const int kbaseB = hi16 << 4;                       // 0 | 16
    const int arow   = (tm << 4) + (lane & 15);
    const int cl     = lane & 15;

    const __bf16* Ap  = A + (long)arow * Kdim + kbaseA;
    const __bf16* Bp0 = Bt + (long)((tn << 6) +  0 + cl) * Kdim + kbaseB;
    const __bf16* Bp1 = Bt + (long)((tn << 6) + 16 + cl) * Kdim + kbaseB;
    const __bf16* Bp2 = Bt + (long)((tn << 6) + 32 + cl) * Kdim + kbaseB;
    const __bf16* Bp3 = Bt + (long)((tn << 6) + 48 + cl) * Kdim + kbaseB;

    v8f acc[4] = {};

    for (int ko = 0; ko < Kdim; ko += 32) {
        // A fragment: contiguous 16 K values per lane (8+8)
        v16bf a;
        {
            v8bf lo = *(const v8bf*)(Ap + ko);
            v8bf hi = *(const v8bf*)(Ap + ko + 8);
#pragma unroll
            for (int i = 0; i < 8; ++i) { a[i] = lo[i]; a[i + 8] = hi[i]; }
        }
        // B fragments: contiguous 16 K values per lane -> distinct SSA values so the
        // scheduler can issue all loads and take partial loadcnt waits per WMMA
        const v16bf b0 = *(const v16bf*)(Bp0 + ko);
        const v16bf b1 = *(const v16bf*)(Bp1 + ko);
        const v16bf b2 = *(const v16bf*)(Bp2 + ko);
        const v16bf b3 = *(const v16bf*)(Bp3 + ko);

        acc[0] = __builtin_amdgcn_wmma_f32_16x16x32_bf16(false, a, false, b0, (short)0, acc[0], false, false);
        acc[1] = __builtin_amdgcn_wmma_f32_16x16x32_bf16(false, a, false, b1, (short)0, acc[1], false, false);
        acc[2] = __builtin_amdgcn_wmma_f32_16x16x32_bf16(false, a, false, b2, (short)0, acc[2], false, false);
        acc[3] = __builtin_amdgcn_wmma_f32_16x16x32_bf16(false, a, false, b3, (short)0, acc[3], false, false);
    }

    // C/D layout: lanes 0-15 hold rows 0..7, lanes 16-31 hold rows 8..15; col = lane&15
    const int orow = (tm << 4) + (hi16 << 3);
#pragma unroll
    for (int j = 0; j < 4; ++j) {
        const int col = (tn << 6) + (j << 4) + cl;
        const float bia = bias ? bias[col] : 0.f;
#pragma unroll
        for (int r = 0; r < 8; ++r) {
            const long idx = (long)(orow + r) * Nout + col;
            float v = acc[j][r] + bia;
            if (res)  v += res[idx];
            if (relu) v = fmaxf(v, 0.f);
            if (outF)  outF[idx] = v;
            if (outBF) outBF[idx] = (__bf16)v;
        }
    }
}

// -------------------------------------------------- edge attention scatter
// one wave per edge; lane = head*4 + chunk; 16-dim dot via shfl_xor reduction
__global__ __launch_bounds__(256) void edge_attn(
    const float* __restrict__ Q, const float* __restrict__ Km,
    const float* __restrict__ V, const int* __restrict__ src,
    const int* __restrict__ dst, float* __restrict__ wV,
    float* __restrict__ z, int E)
{
    const long e = (long)blockIdx.x * 8 + (threadIdx.x >> 5);
    if (e >= E) return;
    const int lane = threadIdx.x & 31;
    const int s = src[e], d = dst[e];
    const int hh = lane >> 2;          // head 0..7
    const int c4 = (lane & 3) * 4;     // 4-elem chunk of the 16-dim head

    const float4 kv = *(const float4*)(Km + (long)s * DMODEL + hh * 16 + c4);
    const float4 qv = *(const float4*)(Q  + (long)d * DMODEL + hh * 16 + c4);
    float part = kv.x * qv.x + kv.y * qv.y + kv.z * qv.z + kv.w * qv.w;
    part += __shfl_xor(part, 1);
    part += __shfl_xor(part, 2);

    float sc = part * 0.25f;                       // 1/sqrt(16)
    sc = fminf(fmaxf(sc, -5.f), 5.f);
    sc = expf(sc);

    const float4 vv = *(const float4*)(V + (long)s * DMODEL + hh * 16 + c4);
    float* wp = wV + (long)d * DMODEL + hh * 16 + c4;
    unsafeAtomicAdd(wp + 0, sc * vv.x);
    unsafeAtomicAdd(wp + 1, sc * vv.y);
    unsafeAtomicAdd(wp + 2, sc * vv.z);
    unsafeAtomicAdd(wp + 3, sc * vv.w);
    if ((lane & 3) == 0) unsafeAtomicAdd(z + (long)d * 8 + hh, sc);
}

// attn = wV / z  ->  bf16 for the Wo GEMM
__global__ __launch_bounds__(256) void attn_div(const float* __restrict__ wV,
                                                const float* __restrict__ z,
                                                __bf16* __restrict__ attnbf, long n) {
    long i = (long)blockIdx.x * 256 + threadIdx.x;
    if (i >= n) return;
    const long node = i >> 7;
    const int hh = (int)((i >> 4) & 7);
    attnbf[i] = (__bf16)(wV[i] / z[node * 8 + hh]);
}

// ---------------------------------------------------------- batchnorm
__global__ __launch_bounds__(256) void bn_stats(const float* __restrict__ x,
                                                float* __restrict__ sum,
                                                float* __restrict__ sumsq,
                                                int Nrows, int rpb) {
    __shared__ float ls[DMODEL], lq[DMODEL];
    const int t = threadIdx.x;
    if (t < DMODEL) { ls[t] = 0.f; lq[t] = 0.f; }
    __syncthreads();
    const int col = t & (DMODEL - 1);
    const long r0 = (long)blockIdx.x * rpb;
    float s = 0.f, q = 0.f;
    for (int r = (t >> 7); r < rpb; r += 2) {
        const long row = r0 + r;
        if (row < Nrows) {
            const float v = x[row * DMODEL + col];
            s += v; q += v * v;
        }
    }
    atomicAdd(&ls[col], s);      // ds_add_f32
    atomicAdd(&lq[col], q);
    __syncthreads();
    if (t < DMODEL) {
        unsafeAtomicAdd(&sum[t],   ls[t]);
        unsafeAtomicAdd(&sumsq[t], lq[t]);
    }
}

__global__ __launch_bounds__(256) void bn_apply(const float* __restrict__ x,
    const float* __restrict__ sum, const float* __restrict__ sumsq,
    const float* __restrict__ g, const float* __restrict__ be,
    float* __restrict__ outF, __bf16* __restrict__ outBF, long n, float invN) {
    long i = (long)blockIdx.x * 256 + threadIdx.x;
    if (i >= n) return;
    const int col = (int)(i & (DMODEL - 1));
    const float mu  = sum[col] * invN;
    const float var = fmaxf(sumsq[col] * invN - mu * mu, 0.f);
    const float rs  = rsqrtf(var + 1e-5f);
    const float v   = (x[i] - mu) * rs * g[col] + be[col];
    if (outF)  outF[i] = v;
    if (outBF) outBF[i] = (__bf16)v;
}

// ------------------------------------------------------------- launcher
extern "C" void kernel_launch(void* const* d_in, const int* in_sizes, int n_in,
                              void* d_out, int out_size, void* d_ws, size_t ws_size,
                              hipStream_t stream) {
    const float* h   = (const float*)d_in[0];
    const int*   src = (const int*)d_in[1];
    const int*   dst = (const int*)d_in[2];
    const float* Wq  = (const float*)d_in[3];
    const float* Wk  = (const float*)d_in[4];
    const float* Wv  = (const float*)d_in[5];
    const float* Wo  = (const float*)d_in[6];
    const float* bo  = (const float*)d_in[7];
    const float* W1  = (const float*)d_in[8];
    const float* b1  = (const float*)d_in[9];
    const float* W2  = (const float*)d_in[10];
    const float* b2  = (const float*)d_in[11];
    const float* g1  = (const float*)d_in[12];
    const float* be1 = (const float*)d_in[13];
    const float* g2  = (const float*)d_in[14];
    const float* be2 = (const float*)d_in[15];
    float* out = (float*)d_out;

    const int N = in_sizes[0] / DMODEL;     // 50000 (multiple of 16)
    const int E = in_sizes[1];              // 800000

    // ---- workspace layout (zero-needing buffers first & contiguous) ----
    char* wsb = (char*)d_ws;
    size_t off = 0;
    auto alloc = [&](size_t bytes) -> void* {
        void* p = wsb + off;
        off += (bytes + 255) & ~(size_t)255;
        return p;
    };
    float* wV    = (float*)alloc((size_t)N * DMODEL * 4);
    float* zsum  = (float*)alloc((size_t)N * 8 * 4);
    float* stats = (float*)alloc(4 * DMODEL * 4);   // sum1|sumsq1|sum2|sumsq2
    const size_t zero_bytes = off;

    float* Qb = (float*)alloc((size_t)N * DMODEL * 4);
    float* Kb = (float*)alloc((size_t)N * DMODEL * 4);
    float* Vb = (float*)alloc((size_t)N * DMODEL * 4);
    __bf16* hbf    = (__bf16*)alloc((size_t)N * DMODEL * 2);
    __bf16* attnbf = (__bf16*)alloc((size_t)N * DMODEL * 2);
    __bf16* xbn1bf = (__bf16*)alloc((size_t)N * DMODEL * 2);
    __bf16* hid    = (__bf16*)alloc((size_t)N * 2 * DMODEL * 2);
    __bf16* Wqt = (__bf16*)alloc(DMODEL * DMODEL * 2);
    __bf16* Wkt = (__bf16*)alloc(DMODEL * DMODEL * 2);
    __bf16* Wvt = (__bf16*)alloc(DMODEL * DMODEL * 2);
    __bf16* Wot = (__bf16*)alloc(DMODEL * DMODEL * 2);
    __bf16* W1t = (__bf16*)alloc(DMODEL * 2 * DMODEL * 2);
    __bf16* W2t = (__bf16*)alloc(2 * DMODEL * DMODEL * 2);

    // buffer aliasing after edge phase consumes Q/K/V
    float* x1   = Qb;   // h + attn@Wo+bo
    float* xbn1 = Kb;   // BN1 output (f32, residual-2 source)
    float* x2r  = Vb;   // xbn1 + FFN output

    const long hn = (long)N * DMODEL;
    const long zn = (long)(zero_bytes / 4);

    // 1) zero accumulators / stats
    zero_f32<<<(unsigned)((zn + 255) / 256), 256, 0, stream>>>((float*)wsb, zn);

    // 2) bf16 conversions / weight transposes
    cvt_bf16<<<(unsigned)((hn + 255) / 256), 256, 0, stream>>>(h, hbf, hn);
    transpose_cvt<<<64, 256, 0, stream>>>(Wq, Wqt, DMODEL, DMODEL);
    transpose_cvt<<<64, 256, 0, stream>>>(Wk, Wkt, DMODEL, DMODEL);
    transpose_cvt<<<64, 256, 0, stream>>>(Wv, Wvt, DMODEL, DMODEL);
    transpose_cvt<<<64, 256, 0, stream>>>(Wo, Wot, DMODEL, DMODEL);
    transpose_cvt<<<128, 256, 0, stream>>>(W1, W1t, DMODEL, 2 * DMODEL);
    transpose_cvt<<<128, 256, 0, stream>>>(W2, W2t, 2 * DMODEL, DMODEL);

    // 3) Q,K,V projections (WMMA)
    const int tiles128 = (N / 16) * (DMODEL / 64);
    const unsigned g128 = (unsigned)((tiles128 + 7) / 8);
    gemm_wmma_bf16<<<g128, 256, 0, stream>>>(hbf, Wqt, nullptr, nullptr, Qb, nullptr, N, DMODEL, DMODEL, 0);
    gemm_wmma_bf16<<<g128, 256, 0, stream>>>(hbf, Wkt, nullptr, nullptr, Kb, nullptr, N, DMODEL, DMODEL, 0);
    gemm_wmma_bf16<<<g128, 256, 0, stream>>>(hbf, Wvt, nullptr, nullptr, Vb, nullptr, N, DMODEL, DMODEL, 0);

    // 4) edge attention scatter (f32 hardware atomics into L2)
    edge_attn<<<(unsigned)((E + 7) / 8), 256, 0, stream>>>(Qb, Kb, Vb, src, dst, wV, zsum, E);

    // 5) normalize attention
    attn_div<<<(unsigned)((hn + 255) / 256), 256, 0, stream>>>(wV, zsum, attnbf, hn);

    // 6) output projection + residual-1 (WMMA)
    gemm_wmma_bf16<<<g128, 256, 0, stream>>>(attnbf, Wot, bo, h, x1, nullptr, N, DMODEL, DMODEL, 0);

    // 7) BN1
    const int rpb = (N + 255) / 256;
    bn_stats<<<256, 256, 0, stream>>>(x1, stats, stats + DMODEL, N, rpb);
    bn_apply<<<(unsigned)((hn + 255) / 256), 256, 0, stream>>>(
        x1, stats, stats + DMODEL, g1, be1, xbn1, xbn1bf, hn, 1.f / (float)N);

    // 8) FFN: relu(x@W1+b1) @ W2 + b2, residual-2 (WMMA)
    const int tiles256 = (N / 16) * (2 * DMODEL / 64);
    gemm_wmma_bf16<<<(unsigned)((tiles256 + 7) / 8), 256, 0, stream>>>(
        xbn1bf, W1t, b1, nullptr, nullptr, hid, N, 2 * DMODEL, DMODEL, 1);
    gemm_wmma_bf16<<<g128, 256, 0, stream>>>(
        hid, W2t, b2, xbn1, x2r, nullptr, N, DMODEL, 2 * DMODEL, 0);

    // 9) BN2 -> output
    bn_stats<<<256, 256, 0, stream>>>(x2r, stats + 2 * DMODEL, stats + 3 * DMODEL, N, rpb);
    bn_apply<<<(unsigned)((hn + 255) / 256), 256, 0, stream>>>(
        x2r, stats + 2 * DMODEL, stats + 3 * DMODEL, g2, be2, out, nullptr, hn, 1.f / (float)N);
}
